// ImageToLatexModel_57552561767135
// MI455X (gfx1250) — matrix operand order repaired and checked by reading, not compiled, not run on hardware
//
#include <hip/hip_runtime.h>
#include <hip/hip_bf16.h>
#include <math.h>

// ---------------------------------------------------------------------------
// Types for WMMA fragments (CDNA5 / gfx1250, wave32)
// ---------------------------------------------------------------------------
typedef __attribute__((ext_vector_type(16))) _Float16 v16h;
typedef __attribute__((ext_vector_type(8)))  _Float16 v8h;
typedef __attribute__((ext_vector_type(8)))  float    v8f;

// Model dims
#define BB   8
#define HIMG 160
#define WIMG 800
#define VV   600
#define EE   256
#define DENC 512
#define DDEC 512
#define AA   256
#define TT   128
#define PP   2000   // 20 * 100 final feature positions
#define HF   20
#define WF   100

// ---------------------------------------------------------------------------
// Layer 0: Cin=1 direct conv (tiny), f32 in -> f16 out, ReLU
// ---------------------------------------------------------------------------
__global__ void conv0_kernel(const float* __restrict__ img,
                             const float* __restrict__ w,
                             const float* __restrict__ bias,
                             _Float16* __restrict__ dst)
{
    long long idx = (long long)blockIdx.x * blockDim.x + threadIdx.x;
    const long long total = (long long)BB * 64 * HIMG * WIMG;
    if (idx >= total) return;
    int x = (int)(idx % WIMG);
    int y = (int)((idx / WIMG) % HIMG);
    int o = (int)((idx / ((long long)WIMG * HIMG)) % 64);
    int b = (int)(idx / ((long long)WIMG * HIMG * 64));
    float s = bias[o];
    #pragma unroll
    for (int ky = 0; ky < 3; ++ky) {
        int yy = y + ky - 1;
        if (yy < 0 || yy >= HIMG) continue;
        #pragma unroll
        for (int kx = 0; kx < 3; ++kx) {
            int xx = x + kx - 1;
            if (xx < 0 || xx >= WIMG) continue;
            s += img[((long long)b * HIMG + yy) * WIMG + xx] * w[o * 9 + ky * 3 + kx];
        }
    }
    s = fmaxf(s, 0.0f);
    dst[idx] = (_Float16)s;
}

// ---------------------------------------------------------------------------
// Weight transform: OIHW f32 -> [tap][Cout][Cin] f16
// ---------------------------------------------------------------------------
__global__ void conv_w_to_f16(const float* __restrict__ w,
                              _Float16* __restrict__ wt,
                              int Cout, int Cin, int taps)
{
    long long idx = (long long)blockIdx.x * blockDim.x + threadIdx.x;
    long long total = (long long)taps * Cout * Cin;
    if (idx >= total) return;
    int i = (int)(idx % Cin);
    int o = (int)((idx / Cin) % Cout);
    int tap = (int)(idx / ((long long)Cin * Cout));
    wt[idx] = (_Float16)w[((long long)o * Cin + i) * taps + tap];
}

__global__ void cast_f16_kernel(const float* __restrict__ x,
                                _Float16* __restrict__ y, long long n)
{
    long long idx = (long long)blockIdx.x * blockDim.x + threadIdx.x;
    if (idx < n) y[idx] = (_Float16)x[idx];
}

// ---------------------------------------------------------------------------
// WMMA conv: 3x3 conv as 9 shifted GEMMs (1 for 1x1). K = Cin (mult of 32).
// One wave -> 32 Cout x 16 x-positions tile (two WMMAs share one B fragment,
// halving streamed-activation bytes per FLOP). Block = 8 waves along x.
// src/dst: (B, C, H, W) f16;  wt: [tap][Cout][Cin] f16;  bias f32.
// ---------------------------------------------------------------------------
__global__ void __launch_bounds__(256) conv_wmma_kernel(
    const _Float16* __restrict__ src, const _Float16* __restrict__ wt,
    const float* __restrict__ bias, _Float16* __restrict__ dst,
    int Cin, int Cout, int H, int W, int taps, int relu)
{
    const int lane = threadIdx.x & 31;
    const int wave = threadIdx.x >> 5;
    const int x0 = (blockIdx.x * 8 + wave) * 16;
    if (x0 >= W) return;                      // wave-uniform
    const int b  = blockIdx.y / H;
    const int y  = blockIdx.y % H;
    const int m0 = blockIdx.z * 32;
    const int n  = lane & 15;
    const int hi = lane >> 4;
    const long long HW = (long long)H * W;
    const int pad = (taps == 9) ? 1 : 0;

    v8f acc0 = {};
    v8f acc1 = {};
    for (int tap = 0; tap < taps; ++tap) {
        const int ky = tap / 3;
        const int kx = tap % 3;
        const int yy = y + ky - pad;
        if (yy < 0 || yy >= H) continue;      // block-uniform
        const int x = x0 + n + kx - pad;
        const bool xin = (x >= 0) && (x < W);
        // A fragment sources: weight rows for out-channels m0+n and m0+16+n
        const _Float16* __restrict__ wrow0 =
            wt + ((long long)tap * Cout + (m0 + n)) * Cin;
        const _Float16* __restrict__ wrow1 = wrow0 + (long long)16 * Cin;
        // B fragment source base (cin = k0 + hi*16 + t, stride HW)
        const long long sbase =
            (((long long)b * Cin + hi * 16) * H + yy) * (long long)W + x;
        for (int k0 = 0; k0 < Cin; k0 += 32) {
            // B fragment (shared by both WMMAs)
            v16h bm;
            {
                const long long base = sbase + (long long)k0 * HW;
                #pragma unroll
                for (int t = 0; t < 16; ++t)
                    bm[t] = xin ? src[base + (long long)t * HW] : (_Float16)0.0f;
            }
            // A fragments (contiguous K: k = hi*16 + t, matching B)
            v16h a0, a1;
            {
                const v8h lo0  = *(const v8h*)(wrow0 + k0 + hi * 16);
                const v8h hi0  = *(const v8h*)(wrow0 + k0 + hi * 16 + 8);
                const v8h lo1  = *(const v8h*)(wrow1 + k0 + hi * 16);
                const v8h hi1  = *(const v8h*)(wrow1 + k0 + hi * 16 + 8);
                #pragma unroll
                for (int t = 0; t < 8; ++t) {
                    a0[t] = lo0[t]; a0[8 + t] = hi0[t];
                    a1[t] = lo1[t]; a1[8 + t] = hi1[t];
                }
            }
            acc0 = __builtin_amdgcn_wmma_f32_16x16x32_f16(
                false, a0, false, bm, (short)0, acc0, false, false);
            acc1 = __builtin_amdgcn_wmma_f32_16x16x32_f16(
                false, a1, false, bm, (short)0, acc1, false, false);
        }
    }
    // D layout: VGPR r, lanes 0-15 -> M=r, lanes 16-31 -> M=r+8; N = lane&15
    const int xo = x0 + n;
    if (xo < W) {
        #pragma unroll
        for (int r = 0; r < 8; ++r) {
            const int o0 = m0 + r + 8 * hi;
            const int o1 = o0 + 16;
            float v0 = acc0[r] + bias[o0];
            float v1 = acc1[r] + bias[o1];
            if (relu) { v0 = fmaxf(v0, 0.0f); v1 = fmaxf(v1, 0.0f); }
            dst[(((long long)b * Cout + o0) * H + y) * (long long)W + xo] = (_Float16)v0;
            dst[(((long long)b * Cout + o1) * H + y) * (long long)W + xo] = (_Float16)v1;
        }
    }
}

// ---------------------------------------------------------------------------
// 2x2 maxpool on f16 (B,C,H,W) -> (B,C,H/2,W/2)
// ---------------------------------------------------------------------------
__global__ void maxpool_kernel(const _Float16* __restrict__ src,
                               _Float16* __restrict__ dst,
                               int C, int H, int W)
{
    const int Ho = H / 2, Wo = W / 2;
    long long idx = (long long)blockIdx.x * blockDim.x + threadIdx.x;
    long long total = (long long)BB * C * Ho * Wo;
    if (idx >= total) return;
    int x = (int)(idx % Wo);
    int y = (int)((idx / Wo) % Ho);
    int c = (int)((idx / ((long long)Wo * Ho)) % C);
    int b = (int)(idx / ((long long)Wo * Ho * C));
    long long base = (((long long)b * C + c) * H + 2 * y) * (long long)W + 2 * x;
    float v0 = (float)src[base],         v1 = (float)src[base + 1];
    float v2 = (float)src[base + W],     v3 = (float)src[base + W + 1];
    dst[idx] = (_Float16)fmaxf(fmaxf(v0, v1), fmaxf(v2, v3));
}

// ---------------------------------------------------------------------------
// Positional encoding + NCHW -> (B,P,D); writes f32 and f16 copies
// ---------------------------------------------------------------------------
__global__ void posenc_kernel(const _Float16* __restrict__ x,
                              float* __restrict__ enc,
                              _Float16* __restrict__ ench)
{
    long long idx = (long long)blockIdx.x * blockDim.x + threadIdx.x;
    const long long total = (long long)BB * PP * DENC;
    if (idx >= total) return;
    int d = (int)(idx % DENC);
    int p = (int)((idx / DENC) % PP);
    int b = (int)(idx / ((long long)DENC * PP));
    int yy = p / WF, xx = p % WF;
    const float kln = logf(10000.0f) / 256.0f;
    float pe;
    if (d < 256) {
        int j = d >> 1;
        float div = expf(-(float)(2 * j) * kln);
        float ang = (float)yy * div;
        pe = (d & 1) ? cosf(ang) : sinf(ang);
    } else {
        int c = d - 256;
        int j = c >> 1;
        float div = expf(-(float)(2 * j) * kln);
        float ang = (float)xx * div;
        pe = (c & 1) ? cosf(ang) : sinf(ang);
    }
    float v = (float)x[(((long long)b * DENC + d) * HF + yy) * WF + xx] + pe;
    enc[idx] = v;
    ench[idx] = (_Float16)v;
}

__global__ void mean_kernel(const float* __restrict__ enc, float* __restrict__ mean)
{
    int idx = blockIdx.x * blockDim.x + threadIdx.x;  // 8*512
    if (idx >= BB * DENC) return;
    int b = idx >> 9, d = idx & 511;
    const float* e = enc + (long long)b * PP * DENC + d;
    float s = 0.0f;
    for (int p = 0; p < PP; ++p) s += e[(long long)p * DENC];
    mean[idx] = s * (1.0f / (float)PP);
}

// ---------------------------------------------------------------------------
// WMMA GEMM: C[M,N] = A[M,K](f16,row-major) * W[N,K]^T + bias[N]   (enc_proj)
// One wave -> 32 M-rows x 16 N-cols (two WMMAs share one W fragment).
// ---------------------------------------------------------------------------
__global__ void __launch_bounds__(256) gemm_wmma_nt(
    const _Float16* __restrict__ A, const _Float16* __restrict__ Wm,
    const float* __restrict__ bias, float* __restrict__ C,
    int M, int N, int K)
{
    const int lane = threadIdx.x & 31;
    const int wave = threadIdx.x >> 5;
    const int n0 = (blockIdx.x * 8 + wave) * 16;
    if (n0 >= N) return;
    const int m0 = blockIdx.y * 32;
    const int n  = lane & 15;
    const int hi = lane >> 4;
    const _Float16* __restrict__ arow0 = A  + (long long)(m0 + n) * K;
    const _Float16* __restrict__ arow1 = arow0 + (long long)16 * K;
    const _Float16* __restrict__ wrow  = Wm + (long long)(n0 + n) * K;
    v8f acc0 = {};
    v8f acc1 = {};
    for (int k0 = 0; k0 < K; k0 += 32) {
        v16h bm;
        {
            const v8h b0 = *(const v8h*)(wrow + k0 + hi * 8);
            const v8h b1 = *(const v8h*)(wrow + k0 + 16 + hi * 8);
            #pragma unroll
            for (int t = 0; t < 8; ++t) { bm[t] = b0[t]; bm[8 + t] = b1[t]; }
        }
        v16h a0, a1;
        {
            const v8h lo0 = *(const v8h*)(arow0 + k0 + hi * 8);
            const v8h hi0 = *(const v8h*)(arow0 + k0 + 16 + hi * 8);
            const v8h lo1 = *(const v8h*)(arow1 + k0 + hi * 8);
            const v8h hi1 = *(const v8h*)(arow1 + k0 + 16 + hi * 8);
            #pragma unroll
            for (int t = 0; t < 8; ++t) {
                a0[t] = lo0[t]; a0[8 + t] = hi0[t];
                a1[t] = lo1[t]; a1[8 + t] = hi1[t];
            }
        }
        acc0 = __builtin_amdgcn_wmma_f32_16x16x32_f16(
            false, a0, false, bm, (short)0, acc0, false, false);
        acc1 = __builtin_amdgcn_wmma_f32_16x16x32_f16(
            false, a1, false, bm, (short)0, acc1, false, false);
    }
    const int col = n0 + n;
    #pragma unroll
    for (int r = 0; r < 8; ++r) {
        const int row0 = m0 + r + 8 * hi;
        C[(long long)row0 * N + col]        = acc0[r] + bias[col];
        C[(long long)(row0 + 16) * N + col] = acc1[r] + bias[col];
    }
}

// ---------------------------------------------------------------------------
// Decoder scalar kernels (batch = 8, precision-sensitive recurrence)
// ---------------------------------------------------------------------------
__device__ __forceinline__ float sigf(float x) { return 1.0f / (1.0f + expf(-x)); }

// Y[b,j] = act( X[b,:] . W[j,:] + bias[j] )  ; grid=(B, ceil(N/256))
__global__ void linear_kernel(const float* __restrict__ X, const float* __restrict__ W,
                              const float* __restrict__ bias, float* __restrict__ Y,
                              int K, int N, int act)
{
    int b = blockIdx.x;
    int j = blockIdx.y * 256 + threadIdx.x;
    if (j >= N) return;
    const float* xr = X + (long long)b * K;
    const float* wr = W + (long long)j * K;
    float s = bias[j];
    for (int k = 0; k < K; ++k) s += xr[k] * wr[k];
    if (act == 1) s = sigf(s);
    Y[b * N + j] = s;
}

// Fused: dec[b,:] = h@W_dec^T + b_dec (staged in LDS), then
// e[b,p] = sum_a w_full[a]*tanh(enc_proj[b,p,a] + dec[b,a]) + b_full
// grid = (B, 4), block = 256; each block covers 500 p's.
__global__ void att_dec_e_kernel(const float* __restrict__ h,
                                 const float* __restrict__ W_dec_att,
                                 const float* __restrict__ b_dec_att,
                                 const float* __restrict__ enc_proj,
                                 const float* __restrict__ w_full,
                                 const float* __restrict__ b_full,
                                 float* __restrict__ e)
{
    __shared__ float sdec[AA];
    __shared__ float swf[AA];
    const int b = blockIdx.x, tid = threadIdx.x;
    {   // one dec element per thread (AA == 256 == blockDim)
        const float* hr = h + b * DDEC;
        const float* wr = W_dec_att + (long long)tid * DDEC;
        float s = b_dec_att[tid];
        for (int k = 0; k < DDEC; ++k) s += hr[k] * wr[k];
        sdec[tid] = s;
        swf[tid] = w_full[tid];
    }
    __syncthreads();
    const int pbase = blockIdx.y * 500;
    const int pend  = pbase + 500;
    for (int p = pbase + tid; p < pend; p += 256) {
        const float* ep = enc_proj + ((long long)b * PP + p) * AA;
        float s = 0.0f;
        for (int a = 0; a < AA; ++a) s += swf[a] * tanhf(ep[a] + sdec[a]);
        e[b * PP + p] = s + b_full[0];
    }
}

// softmax over P then ctx[b,d] = sum_p alpha * enc[b,p,d]; one block per b
__global__ void softmax_ctx_kernel(const float* __restrict__ e,
                                   const float* __restrict__ enc,
                                   float* __restrict__ alpha,
                                   float* __restrict__ ctx)
{
    __shared__ float red[256];
    int b = blockIdx.x, tid = threadIdx.x;
    const float* eb = e + b * PP;
    float mx = -1e30f;
    for (int p = tid; p < PP; p += 256) mx = fmaxf(mx, eb[p]);
    red[tid] = mx; __syncthreads();
    for (int s = 128; s > 0; s >>= 1) {
        if (tid < s) red[tid] = fmaxf(red[tid], red[tid + s]);
        __syncthreads();
    }
    mx = red[0]; __syncthreads();
    float sum = 0.0f;
    for (int p = tid; p < PP; p += 256) {
        float v = expf(eb[p] - mx);
        alpha[b * PP + p] = v;
        sum += v;
    }
    red[tid] = sum; __syncthreads();
    for (int s = 128; s > 0; s >>= 1) {
        if (tid < s) red[tid] += red[tid + s];
        __syncthreads();
    }
    const float inv = 1.0f / red[0];
    __syncthreads();
    for (int d = tid; d < DENC; d += 256) {
        float acc = 0.0f;
        const float* encb = enc + (long long)b * PP * DENC + d;
        const float* al = alpha + b * PP;
        for (int p = 0; p < PP; ++p) acc += al[p] * encb[(long long)p * DENC];
        ctx[b * DENC + d] = acc * inv;
    }
}

// gates[b,j] = [emb(tok), ctx] . W_ih[j] + b_ih[j] + h . W_hh[j] + b_hh[j]
__global__ void gates_kernel(const float* __restrict__ emb, const int* __restrict__ tok,
                             int t, const float* __restrict__ ctx,
                             const float* __restrict__ h,
                             const float* __restrict__ W_ih, const float* __restrict__ W_hh,
                             const float* __restrict__ b_ih, const float* __restrict__ b_hh,
                             float* __restrict__ gates)
{
    int b = blockIdx.x;
    int j = blockIdx.y * 256 + threadIdx.x;  // < 2048
    int token = tok[b * TT + t];
    const float* er  = emb + (long long)token * EE;
    const float* wih = W_ih + (long long)j * (EE + DENC);
    const float* whh = W_hh + (long long)j * DDEC;
    const float* cb = ctx + b * DENC;
    const float* hb = h + b * DDEC;
    float s = b_ih[j] + b_hh[j];
    for (int k = 0; k < EE; ++k)   s += er[k] * wih[k];
    for (int k = 0; k < DENC; ++k) s += cb[k] * wih[EE + k];
    for (int k = 0; k < DDEC; ++k) s += hb[k] * whh[k];
    gates[b * 4 * DDEC + j] = s;
}

// c = sig(f)*c + sig(i)*tanh(g);  h = sig(o)*tanh(c)   (torch gate order i,f,g,o)
__global__ void lstm_update_kernel(const float* __restrict__ gates,
                                   float* __restrict__ h, float* __restrict__ c)
{
    int idx = blockIdx.x * blockDim.x + threadIdx.x;  // 8*512
    if (idx >= BB * DDEC) return;
    int b = idx >> 9, d = idx & 511;
    const float* g = gates + b * 4 * DDEC;
    float ig = sigf(g[d]);
    float fg = sigf(g[DDEC + d]);
    float gg = tanhf(g[2 * DDEC + d]);
    float og = sigf(g[3 * DDEC + d]);
    float cn = fg * c[idx] + ig * gg;
    c[idx] = cn;
    h[idx] = og * tanhf(cn);
}

// Fused: fgate = sigmoid(h@W_fbeta^T + b_fbeta), staged as fgate*ctx in LDS,
// then score[b,t,v] = [h, fgate*ctx] . W_fc[v] + b_fc[v]
__global__ void scores_fused_kernel(const float* __restrict__ h,
                                    const float* __restrict__ ctx,
                                    const float* __restrict__ W_fbeta,
                                    const float* __restrict__ b_fbeta,
                                    const float* __restrict__ W_fc,
                                    const float* __restrict__ b_fc,
                                    float* __restrict__ out, int t)
{
    __shared__ float sg[DENC];   // fgate * ctx
    const int b = blockIdx.x, tid = threadIdx.x;
    const float* hb = h + b * DDEC;
    for (int d = tid; d < DENC; d += 256) {
        const float* wr = W_fbeta + (long long)d * DDEC;
        float s = b_fbeta[d];
        for (int k = 0; k < DDEC; ++k) s += hb[k] * wr[k];
        sg[d] = sigf(s) * ctx[b * DENC + d];
    }
    __syncthreads();
    const int v = blockIdx.y * 256 + tid;
    if (v < VV) {
        const float* wfc = W_fc + (long long)v * (DDEC + DENC);
        float s = b_fc[v];
        for (int k = 0; k < DDEC; ++k) s += hb[k] * wfc[k];
        for (int k = 0; k < DENC; ++k) s += sg[k] * wfc[DDEC + k];
        out[((long long)b * TT + t) * VV + v] = s;
    }
}

// ---------------------------------------------------------------------------
// Host launcher
// ---------------------------------------------------------------------------
extern "C" void kernel_launch(void* const* d_in, const int* in_sizes, int n_in,
                              void* d_out, int out_size, void* d_ws, size_t ws_size,
                              hipStream_t stream)
{
    (void)in_sizes; (void)n_in; (void)out_size; (void)ws_size;
    const float* images = (const float*)d_in[0];
    const int*   tok    = (const int*)d_in[1];
    const float* convw[8]; const float* convb[8];
    for (int i = 0; i < 8; ++i) { convw[i] = (const float*)d_in[2 + i];
                                  convb[i] = (const float*)d_in[10 + i]; }
    const float* embedding = (const float*)d_in[18];
    const float* W_enc_att = (const float*)d_in[19];
    const float* b_enc_att = (const float*)d_in[20];
    const float* W_dec_att = (const float*)d_in[21];
    const float* b_dec_att = (const float*)d_in[22];
    const float* w_full    = (const float*)d_in[23];
    const float* b_full    = (const float*)d_in[24];
    const float* W_init_h  = (const float*)d_in[25];
    const float* b_init_h  = (const float*)d_in[26];
    const float* W_init_c  = (const float*)d_in[27];
    const float* b_init_c  = (const float*)d_in[28];
    const float* W_ih      = (const float*)d_in[29];
    const float* W_hh      = (const float*)d_in[30];
    const float* b_ih      = (const float*)d_in[31];
    const float* b_hh      = (const float*)d_in[32];
    const float* W_fbeta   = (const float*)d_in[33];
    const float* b_fbeta   = (const float*)d_in[34];
    const float* W_fc      = (const float*)d_in[35];
    const float* b_fc      = (const float*)d_in[36];
    float* out = (float*)d_out;

    // ---- workspace layout (256B aligned slots) ----
    char* ws = (char*)d_ws;
    size_t off = 0;
    auto alloc = [&](size_t bytes) {
        size_t o = off; off += (bytes + 255) & ~(size_t)255; return o;
    };
    const long long ACT_ELEMS = (long long)BB * 64 * HIMG * WIMG;  // largest layer
    size_t oA   = alloc((size_t)ACT_ELEMS * 2);
    size_t oB   = alloc((size_t)ACT_ELEMS * 2);
    // f16 transposed conv weights, layers 1..7
    const int LCin[8]  = {1, 64, 64, 128, 128, 256, 256, DENC};
    const int LCout[8] = {64, 64, 128, 128, 256, 256, DENC, DENC};
    const int LTaps[8] = {9, 9, 9, 9, 9, 9, 9, 1};
    size_t owt[8];
    for (int i = 1; i < 8; ++i)
        owt[i] = alloc((size_t)LTaps[i] * LCout[i] * LCin[i] * 2);
    size_t oWenc   = alloc((size_t)AA * DENC * 2);
    size_t oEnc    = alloc((size_t)BB * PP * DENC * 4);
    size_t oEnch   = alloc((size_t)BB * PP * DENC * 2);
    size_t oEproj  = alloc((size_t)BB * PP * AA * 4);
    size_t oMean   = alloc((size_t)BB * DENC * 4);
    size_t oH      = alloc((size_t)BB * DDEC * 4);
    size_t oC      = alloc((size_t)BB * DDEC * 4);
    size_t oE      = alloc((size_t)BB * PP * 4);
    size_t oAlpha  = alloc((size_t)BB * PP * 4);
    size_t oCtx    = alloc((size_t)BB * DENC * 4);
    size_t oGates  = alloc((size_t)BB * 4 * DDEC * 4);

    _Float16* bufA = (_Float16*)(ws + oA);
    _Float16* bufB = (_Float16*)(ws + oB);
    _Float16* wencH = (_Float16*)(ws + oWenc);
    float* enc   = (float*)(ws + oEnc);
    _Float16* ench = (_Float16*)(ws + oEnch);
    float* eproj = (float*)(ws + oEproj);
    float* meanE = (float*)(ws + oMean);
    float* hS    = (float*)(ws + oH);
    float* cS    = (float*)(ws + oC);
    float* eS    = (float*)(ws + oE);
    float* alphaS= (float*)(ws + oAlpha);
    float* ctxS  = (float*)(ws + oCtx);
    float* gatesS= (float*)(ws + oGates);

    // ---- weight conversions ----
    for (int i = 1; i < 8; ++i) {
        long long n = (long long)LTaps[i] * LCout[i] * LCin[i];
        conv_w_to_f16<<<dim3((unsigned)((n + 255) / 256)), dim3(256), 0, stream>>>(
            convw[i], (_Float16*)(ws + owt[i]), LCout[i], LCin[i], LTaps[i]);
    }
    {
        long long n = (long long)AA * DENC;
        cast_f16_kernel<<<dim3((unsigned)((n + 255) / 256)), dim3(256), 0, stream>>>(
            W_enc_att, wencH, n);
    }

    // ---- CNN encoder ----
    conv0_kernel<<<dim3((unsigned)((ACT_ELEMS + 255) / 256)), dim3(256), 0, stream>>>(
        images, convw[0], convb[0], bufA);

    auto conv = [&](const _Float16* s, _Float16* d, int li, int H, int W, int relu) {
        dim3 g((unsigned)((W + 127) / 128), (unsigned)(BB * H), (unsigned)(LCout[li] / 32));
        conv_wmma_kernel<<<g, dim3(256), 0, stream>>>(
            s, (const _Float16*)(ws + owt[li]), convb[li], d,
            LCin[li], LCout[li], H, W, LTaps[li], relu);
    };
    auto pool = [&](const _Float16* s, _Float16* d, int C, int H, int W) {
        long long n = (long long)BB * C * (H / 2) * (W / 2);
        maxpool_kernel<<<dim3((unsigned)((n + 255) / 256)), dim3(256), 0, stream>>>(
            s, d, C, H, W);
    };

    conv(bufA, bufB, 1, 160, 800, 1);  pool(bufB, bufA, 64, 160, 800);
    conv(bufA, bufB, 2, 80, 400, 1);
    conv(bufB, bufA, 3, 80, 400, 1);   pool(bufA, bufB, 128, 80, 400);
    conv(bufB, bufA, 4, 40, 200, 1);
    conv(bufA, bufB, 5, 40, 200, 1);   pool(bufB, bufA, 256, 40, 200);
    conv(bufA, bufB, 6, 20, 100, 1);
    conv(bufB, bufA, 7, 20, 100, 0);   // 1x1, no relu

    // ---- encoder output: pos-enc, mean, init states, enc_proj ----
    {
        long long n = (long long)BB * PP * DENC;
        posenc_kernel<<<dim3((unsigned)((n + 255) / 256)), dim3(256), 0, stream>>>(
            bufA, enc, ench);
    }
    mean_kernel<<<dim3(16), dim3(256), 0, stream>>>(enc, meanE);
    linear_kernel<<<dim3(BB, 2), dim3(256), 0, stream>>>(meanE, W_init_h, b_init_h, hS, DENC, DDEC, 0);
    linear_kernel<<<dim3(BB, 2), dim3(256), 0, stream>>>(meanE, W_init_c, b_init_c, cS, DENC, DDEC, 0);
    gemm_wmma_nt<<<dim3(2, BB * PP / 32), dim3(256), 0, stream>>>(
        ench, wencH, b_enc_att, eproj, BB * PP, AA, DENC);

    // ---- attention-LSTM decoder (sequential over T, 5 launches/step) ----
    for (int t = 0; t < TT; ++t) {
        att_dec_e_kernel<<<dim3(BB, 4), dim3(256), 0, stream>>>(
            hS, W_dec_att, b_dec_att, eproj, w_full, b_full, eS);
        softmax_ctx_kernel<<<dim3(BB), dim3(256), 0, stream>>>(eS, enc, alphaS, ctxS);
        gates_kernel<<<dim3(BB, 8), dim3(256), 0, stream>>>(
            embedding, tok, t, ctxS, hS, W_ih, W_hh, b_ih, b_hh, gatesS);
        lstm_update_kernel<<<dim3(16), dim3(256), 0, stream>>>(gatesS, hS, cS);
        scores_fused_kernel<<<dim3(BB, 3), dim3(256), 0, stream>>>(
            hS, ctxS, W_fbeta, b_fbeta, W_fc, b_fc, out, t);
    }
}
